// E3ConvLayer_15135464751900
// MI455X (gfx1250) — compile-verified, matrix operand order checked by reading, not run on hardware
//
#include <hip/hip_runtime.h>

// Problem constants (from reference)
#define NROWS 100000
#define MNBR  12
#define CCH   64
#define FF    41
#define NM    (NROWS * MNBR)          // 1,200,000 edges (divisible by 128)
#define INV_SQRT_4PI 0.28209479177387814f

typedef __attribute__((ext_vector_type(16))) _Float16 v16h;
typedef __attribute__((ext_vector_type(8)))  float    v8f;

// ---------------------------------------------------------------------------
// WMMA wave32 fragment addressing (ISA 7.12.2), 16x32 f16 operand:
//   element (rc = A-row / B-col, kp = K mod 32) lives in
//   lane = ((kp>>3)&1)*16 + rc, half j = kp<16 ? kp-base : 8+(kp-16-base),
//   base = ((kp>>3)&1)*8. Packing LDS this way makes operand loads b128s.
// ---------------------------------------------------------------------------
__device__ __forceinline__ int frag_slot(int rc, int kp) {
    const int hl   = (kp >> 3) & 1;
    const int base = hl * 8;
    const int j    = (kp < 16) ? (kp - base) : (8 + (kp - 16 - base));
    return (hl * 16 + rc) * 16 + j;
}

// Branch-free softplus on native transcendentals (inputs are ~N(0,1) scale).
__device__ __forceinline__ float softplus_fast(float x) {
    const float t = __builtin_amdgcn_exp2f(fminf(x, 80.0f) * 1.4426950408889634f);
    return 0.6931471805599453f * __builtin_amdgcn_logf(1.0f + t);
}

// ---------------------------------------------------------------------------
// Kernel 1: per-edge radial-MLP scalar via WMMA.
//   s[e] = (softplus(x_e @ W1 + b1) . w2[:,0] + b2[0]) * INV_SQRT_4PI / M
// One wave per 16-edge tile; 8 waves per 256-thread block.
// Hidden GEMM: M=16 edges, N=48 (41 pad), K=64 (41 pad) -> 6 WMMAs/tile.
// ---------------------------------------------------------------------------
__global__ __launch_bounds__(256)
void edge_mlp_kernel(const float* __restrict__ nbr_fea,
                     const float* __restrict__ w1,
                     const float* __restrict__ b1,
                     const float* __restrict__ w2,
                     const float* __restrict__ b2,
                     float* __restrict__ s_out)
{
    __shared__ __align__(32) _Float16 w1p[3 * 2 * 512];  // B frags, 6 KB
    __shared__ __align__(32) _Float16 xp[8][2 * 512];    // A frags, 16 KB
    __shared__ float lds_w2[48];
    __shared__ float lds_b1[48];

    const int tid  = threadIdx.x;
    const int lane = tid & 31;
    const int wave = tid >> 5;

    // Stage W1 (41x41 row-major [k][n]) directly into B-fragment order.
    for (int idx = tid; idx < 64 * 48; idx += 256) {
        const int k = idx / 48, n = idx % 48;
        const float v = (k < FF && n < FF) ? w1[k * FF + n] : 0.0f;
        w1p[((n >> 4) * 2 + (k >> 5)) * 512 + frag_slot(n & 15, k & 31)] =
            (_Float16)v;
    }
    if (tid < 48) {
        lds_w2[tid] = (tid < FF) ? w2[tid * 9] : 0.0f;  // SH_DIM=9, column 0
        lds_b1[tid] = (tid < FF) ? b1[tid] : 0.0f;
    }

    // Stage this wave's 16-edge tile. The tile is a CONTIGUOUS 656-float span
    // (16*41; 16B-aligned since e0 % 16 == 0): load as 164 float4
    // (global_load_b128, coalesced), scatter into A-fragment order.
    const long e0 = (long)(blockIdx.x * 8 + wave) * 16;
    _Float16* xt = xp[wave];
    {   // zero the wave's packed region first (covers K padding 41..63);
        // same-wave LDS ops are in-order, so zero -> scatter is safe.
        const v16h z = {};
        *(v16h*)&xt[lane * 32]      = z;
        *(v16h*)&xt[lane * 32 + 16] = z;
    }
    const float* tile = nbr_fea + e0 * FF;
    for (int q = lane; q < 164; q += 32) {        // 164 * 4 = 656 floats
        const float4 f = *(const float4*)(tile + q * 4);
        const float fv[4] = {f.x, f.y, f.z, f.w};
#pragma unroll
        for (int c = 0; c < 4; ++c) {
            const int idx = q * 4 + c;
            const int r   = idx / FF;             // const-div -> mul/shift
            const int k   = idx - r * FF;
            xt[(k >> 5) * 512 + frag_slot(r, k & 31)] = (_Float16)fv[c];
        }
    }
    __syncthreads();

    const int hl = lane >> 4;
    const int nl = lane & 15;

    float ps[8];
#pragma unroll
    for (int r = 0; r < 8; ++r) ps[r] = 0.0f;
    const float b2_0 = b2[0];

#pragma unroll
    for (int ntile = 0; ntile < 3; ++ntile) {
        v8f acc = {};
#pragma unroll
        for (int kt = 0; kt < 2; ++kt) {
            const v16h a = *(const v16h*)&xt[kt * 512 + lane * 16];
            const v16h b = *(const v16h*)&w1p[(ntile * 2 + kt) * 512 + lane * 16];
            acc = __builtin_amdgcn_wmma_f32_16x16x32_f16(
                false, a, false, b, (short)0, acc, false, false);
        }
        // D layout: this lane owns column ntile*16+nl for rows hl*8 + r.
        const int   ncol = ntile * 16 + nl;
        const float wc   = lds_w2[ncol];
        const float bb   = lds_b1[ncol];
#pragma unroll
        for (int r = 0; r < 8; ++r)
            ps[r] += softplus_fast(acc[r] + bb) * wc;
    }

    // Reduce over the 16 lanes of each half-wave (different cols, same rows).
#pragma unroll
    for (int r = 0; r < 8; ++r) {
        float v = ps[r];
        v += __shfl_xor(v, 1, 32);
        v += __shfl_xor(v, 2, 32);
        v += __shfl_xor(v, 4, 32);
        v += __shfl_xor(v, 8, 32);
        ps[r] = v;
    }
    if (nl == 0) {                       // lane 0 -> rows 0..7, lane 16 -> 8..15
        const int   mb = hl * 8;
        const float sc = INV_SQRT_4PI / (float)MNBR;
        float4 o0, o1;
        o0.x = (ps[0] + b2_0) * sc;  o0.y = (ps[1] + b2_0) * sc;
        o0.z = (ps[2] + b2_0) * sc;  o0.w = (ps[3] + b2_0) * sc;
        o1.x = (ps[4] + b2_0) * sc;  o1.y = (ps[5] + b2_0) * sc;
        o1.z = (ps[6] + b2_0) * sc;  o1.w = (ps[7] + b2_0) * sc;
        *(float4*)(s_out + e0 + mb)     = o0;   // 16B aligned (e0%16==0)
        *(float4*)(s_out + e0 + mb + 4) = o1;
    }
}

// ---------------------------------------------------------------------------
// Kernel 2: weighted gather-accumulate + final tensor-product GEMM via WMMA.
//   agg[i][c] = sum_m s[i,m] * atom_fea[nbr_idx[i,m]][c]   (L2-resident gather)
//   out[i]    = agg[i] @ (tp_w / sqrt(C))                  (16x64 @ 64x64)
// One 256-thread block per 16 rows; coalesced b128 output through LDS.
// ---------------------------------------------------------------------------
__global__ __launch_bounds__(256)
void gather_tp_kernel(const float* __restrict__ atom_fea,
                      const int*   __restrict__ nbr_idx,
                      const float* __restrict__ s_edge,
                      const float* __restrict__ tp_w,
                      float* __restrict__ out)
{
    __shared__ __align__(32) _Float16 aggp[2 * 512];      // A frags, 2 KB
    __shared__ __align__(32) _Float16 wp[4 * 2 * 512];    // B frags, 8 KB
    __shared__ __align__(16) float    lds_out[16 * 64];   // 4 KB
    __shared__ int   lds_idx[16 * MNBR];                  // 192 edges
    __shared__ float lds_s[16 * MNBR];

    const int tid  = threadIdx.x;
    const int lane = tid & 31;
    const int wave = tid >> 5;
    const int i0   = blockIdx.x * 16;

    // Stage tp_w ([k][n] row-major) scaled by 1/sqrt(64) into B-fragment order.
    for (int idx = tid; idx < 64 * 64; idx += 256) {
        const int k = idx >> 6, n = idx & 63;
        const float v = tp_w[idx] * 0.125f;
        wp[((n >> 4) * 2 + (k >> 5)) * 512 + frag_slot(n & 15, k & 31)] =
            (_Float16)v;
    }
    // Stage this block's 192 edge indices/scales once (coalesced).
    if (tid < 16 * MNBR) {
        lds_idx[tid] = nbr_idx[(long)i0 * MNBR + tid];
        lds_s[tid]   = s_edge[(long)i0 * MNBR + tid];
    }
    __syncthreads();

    // Phase A: each wave accumulates 2 rows; each lane owns channels
    // 2*lane, 2*lane+1. Scatter results straight into A-fragment order.
#pragma unroll
    for (int rr = 0; rr < 2; ++rr) {
        const int row = wave * 2 + rr;
        float a0 = 0.0f, a1 = 0.0f;
        for (int mm = 0; mm < MNBR; ++mm) {
            const int   el = row * MNBR + mm;
            const int   j  = lds_idx[el];
            const float sv = lds_s[el];
            if (mm + 1 < MNBR)  // hide L2 gather latency
                __builtin_prefetch(atom_fea + (long)lds_idx[el + 1] * CCH, 0, 0);
            const float2 f = *(const float2*)(atom_fea + (long)j * CCH + lane * 2);
            a0 += sv * f.x;
            a1 += sv * f.y;
        }
        const int c0 = lane * 2;
        aggp[(c0 >> 5) * 512 + frag_slot(row, c0 & 31)]             = (_Float16)a0;
        aggp[((c0 + 1) >> 5) * 512 + frag_slot(row, (c0 + 1) & 31)] = (_Float16)a1;
    }
    __syncthreads();

    // Phase B: out[16,64] = agg @ W; waves 0..3 take one 16x16 column tile.
    if (wave < 4) {
        const int hl = lane >> 4;
        const int nl = lane & 15;
        v8f acc = {};
#pragma unroll
        for (int kt = 0; kt < 2; ++kt) {
            const v16h a = *(const v16h*)&aggp[kt * 512 + lane * 16];
            const v16h b = *(const v16h*)&wp[(wave * 2 + kt) * 512 + lane * 16];
            acc = __builtin_amdgcn_wmma_f32_16x16x32_f16(
                false, a, false, b, (short)0, acc, false, false);
        }
#pragma unroll
        for (int r = 0; r < 8; ++r)
            lds_out[(hl * 8 + r) * 64 + wave * 16 + nl] = acc[r];
    }
    __syncthreads();

    // Coalesced b128 writeback: 16 rows x 64 ch = 1024 floats = 256 x float4.
    const float4 o = *(const float4*)&lds_out[tid * 4];
    *(float4*)(out + (long)i0 * CCH + tid * 4) = o;
}

// ---------------------------------------------------------------------------
extern "C" void kernel_launch(void* const* d_in, const int* in_sizes, int n_in,
                              void* d_out, int out_size, void* d_ws, size_t ws_size,
                              hipStream_t stream)
{
    const float* atom_fea = (const float*)d_in[0];  // [N, C]
    const float* nbr_fea  = (const float*)d_in[1];  // [N, M, F]
    const int*   nbr_idx  = (const int*)d_in[2];    // [N, M]
    // d_in[3] = pos: unused — only the constant l=0 SH channel couples.
    const float* w1   = (const float*)d_in[4];      // [F, F]
    const float* b1   = (const float*)d_in[5];      // [F]
    const float* w2   = (const float*)d_in[6];      // [F, 9]
    const float* b2   = (const float*)d_in[7];      // [9]
    const float* tp_w = (const float*)d_in[8];      // [C, C]

    float* s_edge = (float*)d_ws;                   // NM floats = 4.8 MB

    // 75000 edge tiles / 8 per block = 9375 blocks.
    edge_mlp_kernel<<<NM / (16 * 8), 256, 0, stream>>>(
        nbr_fea, w1, b1, w2, b2, s_edge);
    // 100000 rows / 16 per block = 6250 blocks.
    gather_tp_kernel<<<NROWS / 16, 256, 0, stream>>>(
        atom_fea, nbr_idx, s_edge, tp_w, (float*)d_out);
}
